// MantenPCDEncoder_87806311399886
// MI455X (gfx1250) — compile-verified, matrix operand order checked by reading, not run on hardware
//
#include <hip/hip_runtime.h>
#include <hip/hip_bf16.h>

// ---- problem constants (from reference) ----
#define CAMS   2
#define BATCH  16
#define HDIM   128
#define WDIM   128
#define NPTS   (CAMS * HDIM * WDIM)   // 32768 points per batch
#define NG     128                    // NUM_GROUPS
#define GS     32                     // GROUP_SIZE
#define C1     128                    // hidden 1
#define C2     512                    // hidden 2
#define C3     384                    // output channels
#define MASK_IDENT (-1000000000.0f)
#define PTS_PER_THR (NPTS / 1024)     // 32

typedef __attribute__((ext_vector_type(16))) __bf16 v16bf;
typedef __attribute__((ext_vector_type(8)))  float  v8f;

__device__ __forceinline__ v8f wmma_bf16(v16bf a, v16bf b, v8f c) {
  // D = A(16x32 bf16) * B(32x16 bf16) + C(16x16 f32)
  return __builtin_amdgcn_wmma_f32_16x16x32_bf16(
      /*neg_a=*/false, a, /*neg_b=*/false, b,
      /*c_mod=*/(short)0, c, /*reuse_a=*/false, /*reuse_b=*/false);
}

// K offset inside a 16x32 bf16 A-fragment (or 32x16 B-fragment) for vector
// element e (0..15) of this lane. kb = 0 for lanes 0-15, 8 for lanes 16-31.
// Layout per ISA 7.12.2: V0..V3 hold K = kb + {0..7}, V4..V7 hold K = 16+kb+{0..7}.
__device__ __forceinline__ int wmma_koff(int e, int kb) {
  int i = e >> 1;                       // dword index 0..7
  return ((i & 4) ? 16 : 0) + kb + ((i & 3) << 1) + (e & 1);
}

// ---------------------------------------------------------------------------
// Kernel A: pcd_obs [cam,B,3,H,W] (+mask) -> pc [B, N, 3]  (N = cam*H*W)
// ---------------------------------------------------------------------------
__global__ void k_build_pc(const float* __restrict__ pcd,
                           const unsigned char* __restrict__ msk,
                           float* __restrict__ pc) {
  int idx = blockIdx.x * blockDim.x + threadIdx.x;
  const int total = CAMS * BATCH * HDIM * WDIM;
  if (idx >= total) return;
  int w  = idx % WDIM;
  int h  = (idx / WDIM) % HDIM;
  int b  = (idx / (WDIM * HDIM)) % BATCH;
  int cm = idx / (WDIM * HDIM * BATCH);
  size_t moff = (((size_t)cm * BATCH + b) * HDIM + h) * WDIM + w;
  bool m = msk[moff] != 0;
  int n = (cm * HDIM + h) * WDIM + w;
  float* dst = pc + ((size_t)b * NPTS + n) * 3;
#pragma unroll
  for (int c = 0; c < 3; ++c) {
    float v = pcd[((((size_t)cm * BATCH + b) * 3 + c) * HDIM + h) * WDIM + w];
    dst[c] = m ? v : MASK_IDENT;
  }
}

// ---------------------------------------------------------------------------
// Kernel B: farthest point sampling. One block per batch, 1024 threads.
// Per-thread min-distance state lives in 32 registers (fully unrolled).
// argmax tie-break = smallest index (matches jnp.argmax first-occurrence).
// ---------------------------------------------------------------------------
__global__ __launch_bounds__(1024) void k_fps(const float* __restrict__ pc,
                                              float* __restrict__ centers) {
  const int b = blockIdx.x;
  const int t = threadIdx.x;
  __shared__ float s_last[3];
  __shared__ float s_val[1024];
  __shared__ int   s_idx[1024];

  float d[PTS_PER_THR];
#pragma unroll
  for (int p = 0; p < PTS_PER_THR; ++p) d[p] = 1e30f;

  int last = 0;
  const float* base = pc + (size_t)b * NPTS * 3;

  for (int g = 0; g < NG; ++g) {
    if (t < 3) {
      float cv = base[(size_t)last * 3 + t];
      s_last[t] = cv;
      centers[((size_t)b * NG + g) * 3 + t] = cv;   // record xyz[last]
    }
    __syncthreads();
    float cx = s_last[0], cy = s_last[1], cz = s_last[2];

    float bv = -3e38f; int bi = 0;
#pragma unroll
    for (int p = 0; p < PTS_PER_THR; ++p) {
      int n = t + (p << 10);
      const float* q = base + (size_t)n * 3;
      float dx = q[0] - cx, dy = q[1] - cy, dz = q[2] - cz;
      float dd = dx * dx + dy * dy + dz * dz;
      d[p] = fminf(d[p], dd);
      if (d[p] > bv) { bv = d[p]; bi = n; }          // strict > keeps first max
    }
    s_val[t] = bv; s_idx[t] = bi;
    __syncthreads();
    for (int s = 512; s > 0; s >>= 1) {
      if (t < s) {
        float v2 = s_val[t + s]; int i2 = s_idx[t + s];
        if (v2 > s_val[t] || (v2 == s_val[t] && i2 < s_idx[t])) {
          s_val[t] = v2; s_idx[t] = i2;
        }
      }
      __syncthreads();
    }
    last = s_idx[0];
    __syncthreads();
  }
}

// ---------------------------------------------------------------------------
// Kernel C: per (batch, group): select 32 nearest points, MLP1 (VALU, K=3),
// MLP2 via WMMA bf16 (32x512x128), ReLU, max-pool over the 32 neighbors.
// One block per group-center (2048 blocks), 1024 threads = 32 waves.
// Wave w owns output columns [16w, 16w+16) of the 512-wide hidden layer.
// ---------------------------------------------------------------------------
__global__ __launch_bounds__(1024) void k_group_mlp(
    const float* __restrict__ pc, const float* __restrict__ centers,
    const float* __restrict__ W1, const float* __restrict__ b1,
    const float* __restrict__ W2, const float* __restrict__ b2,
    float* __restrict__ Hfeat) {
  const int gi = blockIdx.x;          // b*NG + g
  const int bI = gi >> 7;
  const int t  = threadIdx.x;

  __shared__ float s_val[1024];
  __shared__ int   s_idx[1024];
  __shared__ int   s_nbr[GS];
  __shared__ float s_x[GS][3];
  __shared__ float s_h1[GS][C1];      // 16 KB

  const float cx = centers[(size_t)gi * 3 + 0];
  const float cy = centers[(size_t)gi * 3 + 1];
  const float cz = centers[(size_t)gi * 3 + 2];
  const float* base = pc + (size_t)bI * NPTS * 3;

  // squared distances of this thread's 32 points to the center (registers)
  float d[PTS_PER_THR];
#pragma unroll
  for (int p = 0; p < PTS_PER_THR; ++p) {
    int n = t + (p << 10);
    const float* q = base + (size_t)n * 3;
    float dx = q[0] - cx, dy = q[1] - cy, dz = q[2] - cz;
    d[p] = dx * dx + dy * dy + dz * dz;
  }

  // iterative argmin selection of the 32 nearest (set equals top_k set)
  for (int i = 0; i < GS; ++i) {
    float bv = 3e38f; int bi = 0x3fffffff;
#pragma unroll
    for (int p = 0; p < PTS_PER_THR; ++p) {
      int n = t + (p << 10);
      if (d[p] < bv) { bv = d[p]; bi = n; }          // strict < keeps first min
    }
    s_val[t] = bv; s_idx[t] = bi;
    __syncthreads();
    for (int s = 512; s > 0; s >>= 1) {
      if (t < s) {
        float v2 = s_val[t + s]; int i2 = s_idx[t + s];
        if (v2 < s_val[t] || (v2 == s_val[t] && i2 < s_idx[t])) {
          s_val[t] = v2; s_idx[t] = i2;
        }
      }
      __syncthreads();
    }
    int win = s_idx[0];
    if (t == 0) s_nbr[i] = win;
#pragma unroll
    for (int p = 0; p < PTS_PER_THR; ++p)
      if (t + (p << 10) == win) d[p] = 3e38f;        // remove from pool
    __syncthreads();
  }

  // gather neighbors, center-relative
  if (t < GS) {
    int n = s_nbr[t];
    const float* q = base + (size_t)n * 3;
    s_x[t][0] = q[0] - cx; s_x[t][1] = q[1] - cy; s_x[t][2] = q[2] - cz;
  }
  __syncthreads();

  // MLP1: h1[32][128] = relu(x @ W1 + b1)   (K=3 -> plain FMA)
#pragma unroll
  for (int c = 0; c < 4; ++c) {
    int o = t + (c << 10);
    int m = o >> 7, k = o & (C1 - 1);
    float acc = b1[k];
    acc += s_x[m][0] * W1[0 * C1 + k];
    acc += s_x[m][1] * W1[1 * C1 + k];
    acc += s_x[m][2] * W1[2 * C1 + k];
    s_h1[m][k] = fmaxf(acc, 0.0f);
  }
  __syncthreads();

  // MLP2 via WMMA: h2 = relu(h1 @ W2 + b2), then max over the 32 rows.
  const int w  = t >> 5;              // wave id 0..31 -> N-tile
  const int l  = t & 31;
  const int cl = l & 15;              // lane column within tile
  const int kb = (l < 16) ? 0 : 8;    // K-half per A/B fragment layout
  const int n  = (w << 4) + cl;       // global column 0..511
  __builtin_prefetch(&W2[n], 0, 3);

  v8f acc0 = {}; v8f acc1 = {};
#pragma unroll
  for (int k0 = 0; k0 < C1; k0 += 32) {
    v16bf A0, A1, Bf;
#pragma unroll
    for (int e = 0; e < 16; ++e) {
      int K = k0 + wmma_koff(e, kb);
      A0[e] = (__bf16)s_h1[cl][K];           // rows 0..15   (M-tile 0)
      A1[e] = (__bf16)s_h1[cl + 16][K];      // rows 16..31  (M-tile 1)
      Bf[e] = (__bf16)W2[(size_t)K * C2 + n];
    }
    acc0 = wmma_bf16(A0, Bf, acc0);
    acc1 = wmma_bf16(A1, Bf, acc1);
  }

  // epilogue: +bias, ReLU, max over the 16+16 rows this lane-pair covers
  float bias = b2[n];
  float mx = -3e38f;
#pragma unroll
  for (int v = 0; v < 8; ++v) {
    mx = fmaxf(mx, fmaxf(acc0[v] + bias, 0.0f));
    mx = fmaxf(mx, fmaxf(acc1[v] + bias, 0.0f));
  }
  mx = fmaxf(mx, __shfl_xor(mx, 16, 32));    // lanes l and l^16 share column n
  if (l < 16) Hfeat[(size_t)gi * C2 + n] = mx;
}

// ---------------------------------------------------------------------------
// Kernel D: out = Hfeat[2048,512] @ W3[512,384] + b3  (WMMA bf16, f32 acc)
// 256 threads = 8 waves per block, one 16x16 tile per wave. 3072 tiles.
// ---------------------------------------------------------------------------
__global__ __launch_bounds__(256) void k_gemm_out(const float* __restrict__ Hf,
                                                  const float* __restrict__ W3,
                                                  const float* __restrict__ b3,
                                                  float* __restrict__ out) {
  const int t = threadIdx.x;
  const int wv = t >> 5, l = t & 31;
  const int NT = C3 / 16;                      // 24 N-tiles
  int tile = blockIdx.x * 8 + wv;
  if (tile >= (BATCH * NG / 16) * NT) return;  // wave-uniform
  const int mt = tile / NT, nt = tile % NT;
  const int cl = l & 15;
  const int kb = (l < 16) ? 0 : 8;
  const int rowA = (mt << 4) + cl;
  const int colB = (nt << 4) + cl;

  v8f acc = {};
#pragma unroll 4
  for (int k0 = 0; k0 < C2; k0 += 32) {
    v16bf A, Bf;
#pragma unroll
    for (int e = 0; e < 16; ++e) {
      int K = k0 + wmma_koff(e, kb);
      A[e]  = (__bf16)Hf[(size_t)rowA * C2 + K];
      Bf[e] = (__bf16)W3[(size_t)K * C3 + colB];
    }
    acc = wmma_bf16(A, Bf, acc);
  }
  float bias = b3[colB];
  int rbase = (mt << 4) + ((l < 16) ? 0 : 8);
#pragma unroll
  for (int v = 0; v < 8; ++v)
    out[(size_t)(rbase + v) * C3 + colB] = acc[v] + bias;
}

// ---------------------------------------------------------------------------
extern "C" void kernel_launch(void* const* d_in, const int* in_sizes, int n_in,
                              void* d_out, int out_size, void* d_ws, size_t ws_size,
                              hipStream_t stream) {
  // inputs (setup_inputs order): rgb_obs, pcd_obs, pcd_mask, W1,b1,W2,b2,W3,b3
  const float* pcd         = (const float*)d_in[1];
  const unsigned char* msk = (const unsigned char*)d_in[2];
  const float* W1 = (const float*)d_in[3];
  const float* b1 = (const float*)d_in[4];
  const float* W2 = (const float*)d_in[5];
  const float* b2 = (const float*)d_in[6];
  const float* W3 = (const float*)d_in[7];
  const float* b3 = (const float*)d_in[8];
  float* out = (float*)d_out;

  // workspace layout (~10.3 MB)
  float* pc      = (float*)d_ws;                               // [B, N, 3]
  float* centers = pc + (size_t)BATCH * NPTS * 3;              // [B, NG, 3]
  float* Hfeat   = centers + (size_t)BATCH * NG * 3;           // [B*NG, 512]

  const int total = CAMS * BATCH * HDIM * WDIM;
  k_build_pc<<<(total + 255) / 256, 256, 0, stream>>>(pcd, msk, pc);
  k_fps<<<BATCH, 1024, 0, stream>>>(pc, centers);
  k_group_mlp<<<BATCH * NG, 1024, 0, stream>>>(pc, centers, W1, b1, W2, b2, Hfeat);
  const int tiles = (BATCH * NG / 16) * (C3 / 16);             // 3072
  k_gemm_out<<<(tiles + 7) / 8, 256, 0, stream>>>(Hfeat, W3, b3, out);
}
